// PairBiasAttention_5686536700299
// MI455X (gfx1250) — matrix operand order repaired and checked by reading, not compiled, not run on hardware
//
#include <hip/hip_runtime.h>

// ---------------------------------------------------------------------------
// PairBiasAttention for MI455X (gfx1250): bf16 WMMA everywhere, pair tensor
// read exactly once, f32 accumulation + f32 softmax.
// ---------------------------------------------------------------------------

typedef unsigned short u16;
typedef __attribute__((ext_vector_type(16))) __bf16 v16bf;
typedef __attribute__((ext_vector_type(8)))  float  v8f;

#define BATCH 4
#define SEQ   512
#define HEADS 8
#define HDIM  32
#define ADIM  256
#define PDIM  64
#define SCALE 0.17677669529663687f   // 1/sqrt(32)
#define NEGINF (-3.0e38f)

struct Pack32 { uint4 lo, hi; };

__device__ __forceinline__ u16 f2bf(float f) {
  unsigned u = __builtin_bit_cast(unsigned, f);
  unsigned r = 0x7FFFu + ((u >> 16) & 1u);     // round-to-nearest-even
  return (u16)((u + r) >> 16);
}

// A-operand (16x32 bf16, MxK): lane l holds row m=l&15, K chunks
// {b..b+7} and {b+16..b+23}, b = (l>=16)*8.  p points at row_start + kk + b.
__device__ __forceinline__ v16bf loadA16(const u16* p) {
  Pack32 t;
  t.lo = *(const uint4*)(p);
  t.hi = *(const uint4*)(p + 16);
  return __builtin_bit_cast(v16bf, t);
}
// B-operand (32x16 bf16, KxN): lane l holds B^T row n=l&15, K chunks
// {b..b+7} and {b+16..b+23}, b = (l>=16)*8.  p points at row_start + kk + b.
__device__ __forceinline__ v16bf loadB16(const u16* p) {
  Pack32 t;
  t.lo = *(const uint4*)(p);
  t.hi = *(const uint4*)(p + 16);
  return __builtin_bit_cast(v16bf, t);
}

__device__ __forceinline__ v8f wmma_bf16(v16bf a, v16bf b, v8f c) {
  return __builtin_amdgcn_wmma_f32_16x16x32_bf16(
      false, a, false, b, (short)0, c, false, false);
}

// ---------------------------------------------------------------------------
// Kernel 1: prep — atom f32 -> bf16 (row major) and Wq/Wk/Wv/Wo f32 ->
// transposed bf16 (WT[c][f] = W[f][c]) so GEMM B-operands are contiguous.
// ---------------------------------------------------------------------------
__global__ __launch_bounds__(256) void prep_kernel(
    const float* __restrict__ atom,
    const float* __restrict__ Wq, const float* __restrict__ Wk,
    const float* __restrict__ Wv, const float* __restrict__ Wo,
    u16* __restrict__ atomb, u16* __restrict__ wqkvT, u16* __restrict__ woT) {
  const int NA = BATCH * SEQ * ADIM;   // 524288
  const int NW = ADIM * ADIM;          // 65536
  int id = blockIdx.x * 256 + threadIdx.x;   // grid covers NA + 4*NW exactly
  if (id < NA) {
    atomb[id] = f2bf(atom[id]);
  } else {
    int t = id - NA;
    int w = t / NW;              // 0..3 : q,k,v,o
    int r = t - w * NW;
    int c = r >> 8;              // output row  (= col of W)
    int f = r & 255;             // output col  (= row of W)
    const float* W = (w == 0) ? Wq : (w == 1) ? Wk : (w == 2) ? Wv : Wo;
    u16 val = f2bf(W[f * ADIM + c]);
    if (w < 3) wqkvT[w * NW + r] = val;
    else       woT[r] = val;
  }
}

// ---------------------------------------------------------------------------
// Kernel 2: QKV projections.  One wave per 16x16 output tile.
// q,k stored bf16 [B][H][N][D]; v stored transposed bf16 [B][H][D][N].
// ---------------------------------------------------------------------------
__global__ __launch_bounds__(256) void qkv_kernel(
    const u16* __restrict__ atomb, const u16* __restrict__ wqkvT,
    const float* __restrict__ bq, const float* __restrict__ bk,
    const float* __restrict__ bv,
    u16* __restrict__ q, u16* __restrict__ k, u16* __restrict__ v) {
  int wave = blockIdx.x * 8 + (threadIdx.x >> 5);
  int l    = threadIdx.x & 31;
  int half = l >> 4;                   // 0 | 1
  int tn = wave & 15;                  // 16 col tiles
  int tm = (wave >> 4) & 127;          // 128 row tiles
  int w  = wave >> 11;                 // 0=q 1=k 2=v

  const u16* Arow = atomb + (tm * 16 + (l & 15)) * ADIM + half * 8;
  const u16* Brow = wqkvT + w * ADIM * ADIM
                  + (tn * 16 + (l & 15)) * ADIM + half * 8;
  v8f acc = {};
#pragma unroll
  for (int kk = 0; kk < ADIM; kk += 32)
    acc = wmma_bf16(loadA16(Arow + kk), loadB16(Brow + kk), acc);

  const float* bias = (w == 0) ? bq : (w == 1) ? bk : bv;
  u16* dst = (w == 0) ? q : (w == 1) ? k : v;
  int n = tn * 16 + (l & 15);          // output column 0..255
  int h = n >> 5, d = n & 31;
  float badd = bias[n];
  int moff = half * 8;
#pragma unroll
  for (int r = 0; r < 8; ++r) {
    int m = r + moff;
    int t = tm * 16 + m;
    int bb = t >> 9, i = t & 511;
    u16 sv = f2bf(acc[r] + badd);
    if (w < 2) dst[(((bb * HEADS + h) * SEQ) + i) * HDIM + d] = sv;     // [B][H][N][D]
    else       dst[(((bb * HEADS + h) * HDIM) + d) * SEQ + i] = sv;     // [B][H][D][N]
  }
}

// ---------------------------------------------------------------------------
// Kernel 3: pair bias.  One thread per (b,i,j): read 64 contiguous f32 ONCE,
// emit all 8 heads -> bias[B][H][N][N] f32.  (Pair tensor = 256 MB dominates
// total traffic; this kernel is the memory-bound hot path.)
// ---------------------------------------------------------------------------
__global__ __launch_bounds__(256) void pair_bias_kernel(
    const float* __restrict__ pair, const float* __restrict__ Wp,
    const float* __restrict__ bp, float* __restrict__ bias) {
  __shared__ float wp[PDIM * HEADS];   // 512 f32
  __shared__ float bpl[HEADS];
  int t = threadIdx.x;
  wp[t] = Wp[t];
  wp[t + 256] = Wp[t + 256];
  if (t < HEADS) bpl[t] = bp[t];
  __syncthreads();

  long id = (long)blockIdx.x * 256 + t;          // (b,i,j), grid exact
  int j  = (int)(id & 511);
  int i  = (int)((id >> 9) & 511);
  int bb = (int)(id >> 18);
  const float* pr = pair + id * PDIM;

  float acc[HEADS] = {0.f, 0.f, 0.f, 0.f, 0.f, 0.f, 0.f, 0.f};
#pragma unroll
  for (int p = 0; p < PDIM; p += 4) {
    float4 pf = *(const float4*)(pr + p);
#pragma unroll
    for (int h = 0; h < HEADS; ++h)
      acc[h] += pf.x * wp[(p + 0) * HEADS + h] + pf.y * wp[(p + 1) * HEADS + h]
              + pf.z * wp[(p + 2) * HEADS + h] + pf.w * wp[(p + 3) * HEADS + h];
  }
#pragma unroll
  for (int h = 0; h < HEADS; ++h)
    bias[(((long)(bb * HEADS + h) * SEQ) + i) * SEQ + j] = acc[h] + bpl[h];
}

// ---------------------------------------------------------------------------
// Kernel 4: attention.  One 128-thread block (4 waves) per (b, h, 16-row
// i-tile).  D=32 == WMMA K, so each 16x16 score tile is ONE wmma.  Full-row
// softmax in LDS (N=512 fits easily in 320 KB), bf16 P back in LDS, then
// P@V with A from LDS (ds_load_b128) and V^T from global.
// ---------------------------------------------------------------------------
__global__ __launch_bounds__(128) void attn_kernel(
    const u16* __restrict__ q, const u16* __restrict__ k,
    const u16* __restrict__ vT, const float* __restrict__ bias,
    const unsigned char* __restrict__ mask, u16* __restrict__ attnout) {
  __shared__ float S[16][SEQ];         // 32 KB scores
  __shared__ u16   Pb[16][SEQ];        // 16 KB bf16 probabilities
  __shared__ float redmax[16][8];
  __shared__ float redsum[16][8];
  __shared__ float Opart[4][16][HDIM]; // 8 KB per-wave partial outputs

  int tid  = threadIdx.x;
  int wv   = tid >> 5;
  int l    = tid & 31;
  int half = l >> 4;
  int moff = half * 8;

  int blk = blockIdx.x;                // bb*256 + h*32 + it
  int it  = blk & 31;
  int h   = (blk >> 5) & 7;
  int bb  = blk >> 8;
  int i0  = it * 16;

  const u16* qh = q  + (bb * HEADS + h) * SEQ * HDIM;     // [N][32]
  const u16* kh = k  + (bb * HEADS + h) * SEQ * HDIM;     // [N][32]
  const u16* vh = vT + (bb * HEADS + h) * HDIM * SEQ;     // [32][N]
  const float* bh = bias + (long)(bb * HEADS + h) * SEQ * SEQ;
  const unsigned char* mb = mask + bb * SEQ;

  // ---- Phase 1: scores.  Q tile (A) is invariant across j-tiles. ----
  v16bf aQ = loadA16(qh + (i0 + (l & 15)) * HDIM + half * 8);
#pragma unroll 2
  for (int jt = wv * 8; jt < wv * 8 + 8; ++jt) {
    int j0 = jt * 16;
    v16bf bK = loadB16(kh + (j0 + (l & 15)) * HDIM + half * 8);
    v8f acc = {};
    acc = wmma_bf16(aQ, bK, acc);
    int n = l & 15;
    bool valid = mb[j0 + n] != 0;
#pragma unroll
    for (int r = 0; r < 8; ++r) {
      int m = r + moff;
      float sv = valid ? (acc[r] * SCALE + bh[(i0 + m) * SEQ + j0 + n])
                       : NEGINF;
      S[m][j0 + n] = sv;
    }
  }
  __syncthreads();

  // ---- Phase 2: softmax (16 rows x 512; 8 threads per row). ----
  int row = tid >> 3;
  int c0  = (tid & 7) * 64;
  float mx = NEGINF;
#pragma unroll 8
  for (int c = 0; c < 64; ++c) mx = fmaxf(mx, S[row][c0 + c]);
  redmax[row][tid & 7] = mx;
  __syncthreads();
  float rmax = redmax[row][0];
#pragma unroll
  for (int x = 1; x < 8; ++x) rmax = fmaxf(rmax, redmax[row][x]);
  float sum = 0.f;
#pragma unroll 8
  for (int c = 0; c < 64; ++c) {
    float e = __expf(S[row][c0 + c] - rmax);
    sum += e;
    Pb[row][c0 + c] = f2bf(e);
  }
  redsum[row][tid & 7] = sum;
  __syncthreads();

  // ---- Phase 3: O = P @ V.  16 K-steps of 32, split across 4 waves. ----
  v8f accO0 = {}, accO1 = {};
  for (int ks = wv; ks < 16; ks += 4) {
    v16bf aP = loadA16(&Pb[l & 15][ks * 32 + half * 8]);
    v16bf b0 = loadB16(vh + (0 * 16 + (l & 15)) * SEQ + ks * 32 + half * 8);
    v16bf b1 = loadB16(vh + (1 * 16 + (l & 15)) * SEQ + ks * 32 + half * 8);
    accO0 = wmma_bf16(aP, b0, accO0);
    accO1 = wmma_bf16(aP, b1, accO1);
  }
#pragma unroll
  for (int r = 0; r < 8; ++r) {
    int m = r + moff;
    Opart[wv][m][(l & 15)]      = accO0[r];
    Opart[wv][m][16 + (l & 15)] = accO1[r];
  }
  __syncthreads();

  // ---- Epilogue: cross-wave reduce, 1/rowsum, store bf16 attnout. ----
  int r2 = tid >> 3;
  int d0 = (tid & 7) * 4;
  float rs = 0.f;
#pragma unroll
  for (int x = 0; x < 8; ++x) rs += redsum[r2][x];
  float inv = 1.0f / rs;
#pragma unroll
  for (int d = d0; d < d0 + 4; ++d) {
    float o = (Opart[0][r2][d] + Opart[1][r2][d] +
               Opart[2][r2][d] + Opart[3][r2][d]) * inv;
    attnout[((bb * SEQ + i0 + r2) * ADIM) + h * HDIM + d] = f2bf(o);
  }
}

// ---------------------------------------------------------------------------
// Kernel 5: output projection  out = attnout(bf16) @ Wo + bo  (f32 out).
// ---------------------------------------------------------------------------
__global__ __launch_bounds__(256) void outproj_kernel(
    const u16* __restrict__ attnout, const u16* __restrict__ woT,
    const float* __restrict__ bo, float* __restrict__ out) {
  int wave = blockIdx.x * 8 + (threadIdx.x >> 5);
  int l    = threadIdx.x & 31;
  int half = l >> 4;
  int tn = wave & 15;
  int tm = wave >> 4;                  // 0..127
  const u16* Arow = attnout + (tm * 16 + (l & 15)) * ADIM + half * 8;
  const u16* Brow = woT     + (tn * 16 + (l & 15)) * ADIM + half * 8;
  v8f acc = {};
#pragma unroll
  for (int kk = 0; kk < ADIM; kk += 32)
    acc = wmma_bf16(loadA16(Arow + kk), loadB16(Brow + kk), acc);
  int n = tn * 16 + (l & 15);
  float badd = bo[n];
#pragma unroll
  for (int r = 0; r < 8; ++r) {
    int m = r + half * 8;
    out[(tm * 16 + m) * ADIM + n] = acc[r] + badd;
  }
}

// ---------------------------------------------------------------------------
extern "C" void kernel_launch(void* const* d_in, const int* in_sizes, int n_in,
                              void* d_out, int out_size, void* d_ws, size_t ws_size,
                              hipStream_t stream) {
  const float* atom = (const float*)d_in[0];
  const float* pair = (const float*)d_in[1];
  const unsigned char* mask = (const unsigned char*)d_in[2];
  const float* Wq = (const float*)d_in[3];
  const float* bq = (const float*)d_in[4];
  const float* Wk = (const float*)d_in[5];
  const float* bk = (const float*)d_in[6];
  const float* Wv = (const float*)d_in[7];
  const float* bv = (const float*)d_in[8];
  const float* Wp = (const float*)d_in[9];
  const float* bp = (const float*)d_in[10];
  const float* Wo = (const float*)d_in[11];
  const float* bo = (const float*)d_in[12];
  float* out = (float*)d_out;

  char* ws = (char*)d_ws;
  size_t off = 0;
  auto take = [&](size_t bytes) -> void* {
    void* p = ws + off;
    off = (off + bytes + 255) & ~(size_t)255;
    return p;
  };
  const size_t TOK = (size_t)BATCH * SEQ;                    // 2048
  u16* atomb  = (u16*)take(TOK * ADIM * 2);                  // 1 MB
  u16* wqkvT  = (u16*)take((size_t)3 * ADIM * ADIM * 2);     // 384 KB
  u16* woT    = (u16*)take((size_t)ADIM * ADIM * 2);         // 128 KB
  u16* qb     = (u16*)take(TOK * ADIM * 2);                  // 1 MB
  u16* kb     = (u16*)take(TOK * ADIM * 2);                  // 1 MB
  u16* vb     = (u16*)take(TOK * ADIM * 2);                  // 1 MB (transposed per head)
  u16* attnob = (u16*)take(TOK * ADIM * 2);                  // 1 MB
  float* bias = (float*)take((size_t)BATCH * HEADS * SEQ * SEQ * 4); // 33.5 MB

  // 1) precision prep (exact grid: 524288 + 262144 threads)
  prep_kernel<<<3072, 256, 0, stream>>>(atom, Wq, Wk, Wv, Wo, atomb, wqkvT, woT);
  // 2) QKV projections: 3 * 128 * 16 = 6144 waves, 8 waves/block
  qkv_kernel<<<768, 256, 0, stream>>>(atomb, wqkvT, bq, bk, bv, qb, kb, vb);
  // 3) pair bias: one thread per (b,i,j) = 1,048,576 threads
  pair_bias_kernel<<<4096, 256, 0, stream>>>(pair, Wp, bp, bias);
  // 4) attention: one block per (b,h,i-tile) = 4*8*32 = 1024 blocks
  attn_kernel<<<1024, 128, 0, stream>>>(qb, kb, vb, bias, mask, attnob);
  // 5) output projection: 128*16 = 2048 waves
  outproj_kernel<<<256, 256, 0, stream>>>(attnob, woT, bo, out);
}
